// WindowAttention_kv_29832842838659
// MI455X (gfx1250) — matrix operand order repaired and checked by reading, not compiled, not run on hardware
//
#include <hip/hip_runtime.h>

// ---------------------------------------------------------------------------
// CDNA5 (gfx1250) fused window-attention pipeline:
//   v_wmma_f32_16x16x32_bf16 for all GEMMs + TDM (tensor_load_to_lds) staging
// ---------------------------------------------------------------------------

typedef __attribute__((ext_vector_type(16))) __bf16 v16bf;
typedef __attribute__((ext_vector_type(8)))  __bf16 v8bf;
typedef __attribute__((ext_vector_type(8)))  float  v8f;
typedef __attribute__((ext_vector_type(4)))  unsigned int u32x4;
typedef __attribute__((ext_vector_type(8)))  int i32x8;
typedef __attribute__((ext_vector_type(4)))  int i32x4;

#define DIM   192
#define HEADS 6
#define HD    32
#define NTOK  98
#define NPAD  112   // tokens padded to multiple of 16 (7 tiles)
#define KPAD  128   // key dim padded to multiple of 32 for the P*V K-loop
#define NWIN  64

__device__ __forceinline__ v8f zero8() {
  v8f z;
#pragma unroll
  for (int i = 0; i < 8; ++i) z[i] = 0.0f;
  return z;
}

__device__ __forceinline__ v8f wmma_bf16(v16bf a, v16bf b, v8f c) {
  // D = A*B + C, 16x16x32, bf16 inputs, f32 accumulate
  return __builtin_amdgcn_wmma_f32_16x16x32_bf16(false, a, false, b, (short)0, c,
                                                 false, false);
}

// A-matrix fragment (16x32, M=row per lane) from row-major f32 data.
__device__ __forceinline__ v16bf a_frag_f32(const float* row, int k0, int hi,
                                            float scale, bool valid) {
  v16bf a;
  if (valid) {
    const float4* p1 = (const float4*)(row + k0 + 16 * hi);
    const float4* p2 = (const float4*)(row + k0 + 16 * hi + 8);
    float4 xs[4];
    xs[0] = p1[0]; xs[1] = p1[1]; xs[2] = p2[0]; xs[3] = p2[1];
    const float* tf = (const float*)xs;
#pragma unroll
    for (int i = 0; i < 16; ++i) a[i] = (__bf16)(tf[i] * scale);
  } else {
#pragma unroll
    for (int i = 0; i < 16; ++i) a[i] = (__bf16)0.0f;
  }
  return a;
}

// Same but from row-major bf16 data (always valid).
__device__ __forceinline__ v16bf a_frag_bf16(const __bf16* row, int k0, int hi) {
  v8bf c1 = *(const v8bf*)(row + k0 + 16 * hi);
  v8bf c2 = *(const v8bf*)(row + k0 + 16 * hi + 8);
  v16bf a;
#pragma unroll
  for (int i = 0; i < 8; ++i) { a[i] = c1[i]; a[8 + i] = c2[i]; }
  return a;
}

// ---------------------------------------------------------------------------
// Tensor Data Mover: 2D tile load global -> LDS (descriptor per ISA ch.8)
// dim0 = contiguous elements (2B each), dim1 = rows, stride0 = row stride.
// Issued by one wave; completion via s_wait_tensorcnt + workgroup barrier.
// ---------------------------------------------------------------------------
__device__ __forceinline__ void tdm_stage_2d(const __bf16* gsrc, __bf16* lds,
                                             unsigned dim0, unsigned dim1,
                                             unsigned stride0, int wv, int tix,
                                             int nthreads) {
#if defined(__AMDGCN__) && __has_builtin(__builtin_amdgcn_tensor_load_to_lds)
  if (wv == 0) {
    unsigned long long ga = (unsigned long long)(uintptr_t)gsrc;
    unsigned lo = (unsigned)(uintptr_t)lds;   // LDS offset = addr[31:0]
    u32x4 g0;
    g0[0] = 1u;                                   // count=1, user descriptor
    g0[1] = lo;                                   // lds_addr
    g0[2] = (unsigned)ga;                         // global_addr[31:0]
    g0[3] = (unsigned)((ga >> 32) & 0x01FFFFFFull) | (2u << 30);  // type=2
    i32x8 g1;
    g1[0] = 1 << 16;                              // data_size=1 (2 bytes)
    g1[1] = (int)((dim0 & 0xFFFFu) << 16);        // tensor_dim0 lo16
    g1[2] = (int)((dim0 >> 16) | ((dim1 & 0xFFFFu) << 16));   // dim0 hi / dim1 lo
    g1[3] = (int)((dim1 >> 16) | ((dim0 & 0xFFFFu) << 16));   // dim1 hi / tile_dim0
    g1[4] = (int)(dim1 & 0xFFFFu);                // tile_dim1 (tile_dim2 = 0)
    g1[5] = (int)stride0;                         // tensor_dim0_stride lo32
    g1[6] = 0;
    g1[7] = 0;                                    // tensor_dim1_stride = 0 (2D)
    i32x4 g2, g3;
#pragma unroll
    for (int i = 0; i < 4; ++i) { g2[i] = 0; g3[i] = 0; }
#if __clang_major__ >= 23
    i32x8 g4;
#pragma unroll
    for (int i = 0; i < 8; ++i) g4[i] = 0;
    __builtin_amdgcn_tensor_load_to_lds(g0, g1, g2, g3, g4, 0);
#else
    __builtin_amdgcn_tensor_load_to_lds(g0, g1, g2, g3, 0);
#endif
  }
#else
  // Fallback (host pass / toolchains without the TDM builtin)
  int count = (int)(dim0 * dim1);
  for (int i = tix; i < count; i += nthreads)
    lds[i] = gsrc[(i / (int)dim0) * (int)stride0 + (i % (int)dim0)];
#endif
}

__device__ __forceinline__ void tdm_wait(int wv) {
#if defined(__AMDGCN__) && __has_builtin(__builtin_amdgcn_s_wait_tensorcnt)
  if (wv == 0) __builtin_amdgcn_s_wait_tensorcnt(0);
#endif
}

// ---------------------------------------------------------------------------
// Kernel 1: weight transposes to bf16 [n][k] layout + combined bias/mask table
// ---------------------------------------------------------------------------
__global__ void __launch_bounds__(256)
prep_kernel(const float* __restrict__ kv_w, const float* __restrict__ proj_w,
            const float* __restrict__ rpb, const int* __restrict__ rel_idx,
            const float* __restrict__ mask,
            __bf16* __restrict__ kv_wT, __bf16* __restrict__ proj_wT,
            float* __restrict__ biasm) {
  int tid = blockIdx.x * blockDim.x + threadIdx.x;
  int nth = gridDim.x * blockDim.x;
  for (int i = tid; i < 2 * DIM * DIM; i += nth) {         // kv_wT[384][192]
    int n = i / DIM, k = i % DIM;
    kv_wT[i] = (__bf16)kv_w[k * (2 * DIM) + n];
  }
  for (int i = tid; i < DIM * DIM; i += nth) {             // proj_wT[192][192]
    int n = i / DIM, k = i % DIM;
    proj_wT[i] = (__bf16)proj_w[k * DIM + n];
  }
  // biasm[w][h][112][112] = rpb gather + window mask, pads = -1e30
  for (int i = tid; i < NWIN * HEADS * NPAD * NPAD; i += nth) {
    int w = i / (HEADS * NPAD * NPAD);
    int r = i % (HEADS * NPAD * NPAD);
    int h = r / (NPAD * NPAD);
    int r2 = r % (NPAD * NPAD);
    int q = r2 / NPAD, k = r2 % NPAD;
    float v = -1e30f;
    if (q < NTOK && k < NTOK)
      v = rpb[rel_idx[q * NTOK + k] * HEADS + h] + mask[(w * NTOK + q) * NTOK + k];
    biasm[i] = v;
  }
}

// ---------------------------------------------------------------------------
// Kernel 2: KV = skip @ kv_w + kv_b  (per batch; K->[h][key][d], V->[h][d][key])
// ---------------------------------------------------------------------------
__global__ void __launch_bounds__(256)
kv_kernel(const float* __restrict__ skip, const float* __restrict__ kv_bias,
          const __bf16* __restrict__ kv_wT,
          __bf16* __restrict__ kbuf, __bf16* __restrict__ vtbuf) {
  int b = blockIdx.x;
  int lane = threadIdx.x & 31;
  int wv = threadIdx.x >> 5;
  int ln = lane & 15, hi = lane >> 4;

  const float* A = skip + (size_t)b * NTOK * DIM;
  __bf16* kb = kbuf + (size_t)b * HEADS * NPAD * HD;
  __bf16* vb = vtbuf + (size_t)b * HEADS * HD * KPAD;

  // 7 M-tiles x 24 N-tiles = 168 tiles, 21 per wave
  for (int t = wv; t < 7 * 24; t += 8) {
    int m0 = (t / 24) * 16, n0 = (t % 24) * 16;
    int m = m0 + ln;
    bool valid = (m < NTOK);
    const float* arow = A + (size_t)m * DIM;
    v8f acc = zero8();
#pragma unroll
    for (int k0 = 0; k0 < DIM; k0 += 32) {
      v16bf a = a_frag_f32(arow, k0, hi, 1.0f, valid);
      v16bf bm = *(const v16bf*)(kv_wT + (size_t)(n0 + ln) * DIM + k0 + 16 * hi);
      acc = wmma_bf16(a, bm, acc);
    }
    int c = n0 + ln;
    float cb = kv_bias[c];
#pragma unroll
    for (int r = 0; r < 8; ++r) {
      int key = m0 + 8 * hi + r;
      float val = acc[r] + cb;
      __bf16 o = (key < NTOK) ? (__bf16)val : (__bf16)0.0f;
      if (c < DIM) {                 // K part: [h][key][d]
        int h = c >> 5, d = c & 31;
        kb[((size_t)h * NPAD + key) * HD + d] = o;
      } else {                       // V part transposed: [h][d][key]
        int c2 = c - DIM;
        int h = c2 >> 5, d = c2 & 31;
        vb[((size_t)h * HD + d) * KPAD + key] = o;
      }
    }
  }
  // zero V^T pad keys [112,128) so P*V pad columns are well-defined
  for (int i = threadIdx.x; i < HEADS * HD * (KPAD - NPAD); i += 256) {
    int h = i / (HD * 16);
    int rr = i % (HD * 16);
    int d = rr / 16;
    int kk = NPAD + (rr & 15);
    vb[((size_t)h * HD + d) * KPAD + kk] = (__bf16)0.0f;
  }
}

// ---------------------------------------------------------------------------
// Kernel 3: per (batch, head): S = QK^T + bias+mask, softmax, O = P V
//           K / V^T head slabs staged in LDS via the Tensor Data Mover
// ---------------------------------------------------------------------------
__global__ void __launch_bounds__(224)
att_kernel(const float* __restrict__ x_up, const float* __restrict__ biasm,
           const __bf16* __restrict__ kbuf, const __bf16* __restrict__ vtbuf,
           __bf16* __restrict__ obuf) {
  __shared__ float S[7][16][KPAD];                              // 57,344 B
  __shared__ __attribute__((aligned(32))) __bf16 stage[4096];   //  8,192 B

  int b = blockIdx.x, h = blockIdx.y;
  int lane = threadIdx.x & 31, wv = threadIdx.x >> 5;
  int ln = lane & 15, hi = lane >> 4;
  int m0 = wv * 16;

  const float scale = 0.17677669529663689f;  // 32^-0.5
  int m = m0 + ln;
  const float* qrow = x_up + ((size_t)b * NTOK + m) * DIM + h * HD;
  v16bf aq = a_frag_f32(qrow, 0, hi, scale, m < NTOK);

  // --- TDM: K slab [112 keys][32 d] -> LDS ---
  const __bf16* kb = kbuf + ((size_t)b * HEADS + h) * NPAD * HD;
  tdm_stage_2d(kb, stage, HD, NPAD, HD, wv, threadIdx.x, 224);
  tdm_wait(wv);
  __syncthreads();

  const float* bm = biasm + ((size_t)(b & (NWIN - 1)) * HEADS + h) * NPAD * NPAD;

  // ---- S = Q K^T + (bias+mask) : 7 N-tiles, one wmma each (K = 32) ----
#pragma unroll
  for (int n0t = 0; n0t < 7; ++n0t) {
    int n0 = n0t * 16;
    v16bf bk = *(const v16bf*)(stage + (size_t)(n0 + ln) * HD + 16 * hi);
    v8f acc = zero8();
    acc = wmma_bf16(aq, bk, acc);
    int k = n0 + ln;
#pragma unroll
    for (int r = 0; r < 8; ++r) {
      int q = m0 + 8 * hi + r;
      S[wv][8 * hi + r][k] = acc[r] + bm[q * NPAD + k];
    }
  }
  // pad columns 112..127
  for (int i2 = lane; i2 < 16 * 16; i2 += 32)
    S[wv][i2 >> 4][NPAD + (i2 & 15)] = -1e30f;
  __syncthreads();   // all waves done reading K slab -> stage reusable

  // --- TDM: V^T slab [32 d][128 keys] -> LDS (overlapped with softmax) ---
  const __bf16* vbg = vtbuf + ((size_t)b * HEADS + h) * HD * KPAD;
  tdm_stage_2d(vbg, stage, KPAD, HD, KPAD, wv, threadIdx.x, 224);

  // ---- softmax: 2 lanes per row, cross-half reduce with shfl_xor(16) ----
  {
    int row = ln;
    int c0 = hi * 64;
    float mx = -3.0e38f;
    for (int c = 0; c < 64; ++c) mx = fmaxf(mx, S[wv][row][c0 + c]);
    mx = fmaxf(mx, __shfl_xor(mx, 16, 32));
    float sum = 0.0f;
    for (int c = 0; c < 64; ++c) {
      float e = __expf(S[wv][row][c0 + c] - mx);
      S[wv][row][c0 + c] = e;
      sum += e;
    }
    sum += __shfl_xor(sum, 16, 32);
    float inv = 1.0f / sum;
    for (int c = 0; c < 64; ++c) S[wv][row][c0 + c] *= inv;
  }
  tdm_wait(wv);
  __syncthreads();

  // ---- O = P V : 2 N-tiles (32 dims), K-loop 128/32 = 4 ----
  __bf16* ob = obuf + (size_t)b * NPAD * DIM;
#pragma unroll
  for (int nt = 0; nt < 2; ++nt) {
    v8f acc = zero8();
#pragma unroll
    for (int kk = 0; kk < KPAD; kk += 32) {
      v16bf ap = a_frag_f32(&S[wv][ln][0], kk, hi, 1.0f, true);
      v16bf bv = *(const v16bf*)(stage + (size_t)(nt * 16 + ln) * KPAD + kk + 16 * hi);
      acc = wmma_bf16(ap, bv, acc);
    }
    int col = h * HD + nt * 16 + ln;
#pragma unroll
    for (int r = 0; r < 8; ++r) {
      int q = m0 + 8 * hi + r;
      ob[(size_t)q * DIM + col] = (q < NTOK) ? (__bf16)acc[r] : (__bf16)0.0f;
    }
  }
}

// ---------------------------------------------------------------------------
// Kernel 4: out = O @ proj_w + proj_b  (f32 output)
// ---------------------------------------------------------------------------
__global__ void __launch_bounds__(256)
proj_kernel(const __bf16* __restrict__ obuf, const __bf16* __restrict__ proj_wT,
            const float* __restrict__ proj_b, float* __restrict__ out) {
  int b = blockIdx.x;
  int lane = threadIdx.x & 31, wv = threadIdx.x >> 5;
  int ln = lane & 15, hi = lane >> 4;
  const __bf16* A = obuf + (size_t)b * NPAD * DIM;

  for (int t = wv; t < 7 * 12; t += 8) {
    int m0 = (t / 12) * 16, n0 = (t % 12) * 16;
    const __bf16* arow = A + (size_t)(m0 + ln) * DIM;
    v8f acc = zero8();
#pragma unroll
    for (int k0 = 0; k0 < DIM; k0 += 32) {
      v16bf a = a_frag_bf16(arow, k0, hi);
      v16bf bmf = *(const v16bf*)(proj_wT + (size_t)(n0 + ln) * DIM + k0 + 16 * hi);
      acc = wmma_bf16(a, bmf, acc);
    }
    int c = n0 + ln;
    float pb = proj_b[c];
#pragma unroll
    for (int r = 0; r < 8; ++r) {
      int q = m0 + 8 * hi + r;
      if (q < NTOK) out[((size_t)b * NTOK + q) * DIM + c] = acc[r] + pb;
    }
  }
}

// ---------------------------------------------------------------------------
// Host-side launcher
// ---------------------------------------------------------------------------
extern "C" void kernel_launch(void* const* d_in, const int* in_sizes, int n_in,
                              void* d_out, int out_size, void* d_ws, size_t ws_size,
                              hipStream_t stream) {
  (void)in_sizes; (void)n_in; (void)out_size; (void)ws_size;
  const float* skip   = (const float*)d_in[0];
  const float* x_up   = (const float*)d_in[1];
  const float* mask   = (const float*)d_in[2];
  const float* kv_w   = (const float*)d_in[3];
  const float* kv_b   = (const float*)d_in[4];
  const float* proj_w = (const float*)d_in[5];
  const float* proj_b = (const float*)d_in[6];
  const float* rpb    = (const float*)d_in[7];
  const int*   rel_i  = (const int*)d_in[8];
  float* out = (float*)d_out;

  char* ws = (char*)d_ws;
  // workspace carve-up (all offsets 256B-aligned), total = 157,900,800 B
  __bf16* kv_wT   = (__bf16*)(ws + 0);           //    147,456 B [384][192] bf16
  __bf16* proj_wT = (__bf16*)(ws + 147456);      //     73,728 B [192][192] bf16
  float*  biasm   = (float*) (ws + 221184);      // 19,267,584 B [64][6][112][112] f32
  __bf16* kbuf    = (__bf16*)(ws + 19488768);    // 44,040,192 B [1024][6][112][32] bf16
  __bf16* vtbuf   = (__bf16*)(ws + 63528960);    // 50,331,648 B [1024][6][32][128] bf16
  __bf16* obuf    = (__bf16*)(ws + 113860608);   // 44,040,192 B [1024][112][192] bf16

  prep_kernel<<<dim3(512), dim3(256), 0, stream>>>(kv_w, proj_w, rpb, rel_i, mask,
                                                   kv_wT, proj_wT, biasm);
  kv_kernel<<<dim3(1024), dim3(256), 0, stream>>>(skip, kv_b, kv_wT, kbuf, vtbuf);
  att_kernel<<<dim3(1024, 6), dim3(224), 0, stream>>>(x_up, biasm, kbuf, vtbuf, obuf);
  proj_kernel<<<dim3(1024), dim3(256), 0, stream>>>(obuf, proj_wT, proj_b, out);
}
